// MultiHeadedSelfAttention_38869454029041
// MI455X (gfx1250) — matrix operand order repaired and checked by reading, not compiled, not run on hardware
//
#include <hip/hip_runtime.h>
#include <hip/hip_bf16.h>

// MI455X / gfx1250, wave32. bf16 WMMA flash attention, v2:
//  - QKV GEMM: 32x64 tile per wave (8 WMMA per K-step) for fragment reuse
//  - attention: 64-key chunks to amortize softmax reductions

typedef __attribute__((ext_vector_type(16))) __bf16 v16bf;
typedef __attribute__((ext_vector_type(8)))  __bf16 v8bf;
typedef __attribute__((ext_vector_type(8)))  float  v8f;

#define S_LEN 4096
#define D_DIM 768
#define H_NUM 12
#define W_HD  64
#define QKV3  2304   // 3*D

#define WMMA_BF16(A, B, C) \
  __builtin_amdgcn_wmma_f32_16x16x32_bf16(false, (A), false, (B), (short)0, (C), false, false)

// ---------- helpers ----------

// Build a v16bf operand from two 16-byte chunks.
static __device__ __forceinline__ v16bf load_frag16(const __bf16* p0, const __bf16* p1) {
  v8bf lo = *(const v8bf*)p0;
  v8bf hi = *(const v8bf*)p1;
  v16bf r;
#pragma unroll
  for (int i = 0; i < 8; ++i) { r[i] = lo[i]; r[8 + i] = hi[i]; }
  return r;
}

// reductions across the 16 lanes of each half-wave
static __device__ __forceinline__ float rmax16(float v) {
#pragma unroll
  for (int m = 1; m <= 8; m <<= 1) v = fmaxf(v, __shfl_xor(v, m, 32));
  return v;
}
static __device__ __forceinline__ float rsum16(float v) {
#pragma unroll
  for (int m = 1; m <= 8; m <<= 1) v += __shfl_xor(v, m, 32);
  return v;
}

// ---------- conversion kernels ----------

__global__ void conv_x_kernel(const float* __restrict__ x, __bf16* __restrict__ xb, int n) {
  int i = blockIdx.x * 256 + threadIdx.x;
  if (i < n) xb[i] = (__bf16)x[i];
}

// W (D x 3D) fp32 -> Wt (3D x D) bf16 (transposed)
__global__ void conv_w_kernel(const float* __restrict__ w, __bf16* __restrict__ wt) {
  int i = blockIdx.x * 256 + threadIdx.x;
  if (i < D_DIM * QKV3) {
    int k = i / QKV3;
    int n = i % QKV3;
    wt[n * D_DIM + k] = (__bf16)w[i];
  }
}

__global__ void conv_mask_kernel(const float* __restrict__ mask, float* __restrict__ ma) {
  int i = blockIdx.x * 256 + threadIdx.x;
  if (i < S_LEN) ma[i] = -10000.0f * (1.0f - mask[i]);
}

// ---------- QKV projection GEMM ----------
// 32x64 output tile per wave: 2 M-subtiles x 4 N-subtiles, K-loop of 24 steps,
// 8 WMMAs per step with 2 A-frags / 4 B-frags (2-4x fragment reuse).
__global__ void __launch_bounds__(256)
qkv_gemm_kernel(const __bf16* __restrict__ xb, const __bf16* __restrict__ wt,
                const float* __restrict__ bias,
                __bf16* __restrict__ Qs, __bf16* __restrict__ Ks, __bf16* __restrict__ Vt)
{
  const int lane  = threadIdx.x & 31;
  const int gw    = blockIdx.x * 8 + (threadIdx.x >> 5);
  const int tileM = gw / (QKV3 / 64);   // 128 tiles of 32 rows
  const int tileN = gw % (QKV3 / 64);   // 36 tiles of 64 cols
  const int half  = lane >> 4;
  const int l16   = lane & 15;
  const int aoff  = half * 8;           // A layout: K chunks {0..7,16..23} (+8 upper half)
  const int boff  = half * 8;           // B layout: K chunks {0..7,16..23} (+8 upper half)

  const __bf16* arow0 = xb + (size_t)(tileM * 32 + l16) * D_DIM;
  const __bf16* arow1 = arow0 + (size_t)16 * D_DIM;
  const __bf16* brow[4];
#pragma unroll
  for (int j = 0; j < 4; ++j)
    brow[j] = wt + (size_t)(tileN * 64 + j * 16 + l16) * D_DIM;

  v8f acc[2][4] = {};
  for (int k = 0; k < D_DIM; k += 32) {
    const v16bf a0 = load_frag16(arow0 + k + aoff, arow0 + k + 16 + aoff);
    const v16bf a1 = load_frag16(arow1 + k + aoff, arow1 + k + 16 + aoff);
#pragma unroll
    for (int j = 0; j < 4; ++j) {
      const v16bf b = load_frag16(brow[j] + k + boff, brow[j] + k + boff + 16);
      acc[0][j] = WMMA_BF16(a0, b, acc[0][j]);
      acc[1][j] = WMMA_BF16(a1, b, acc[1][j]);
    }
  }

  // epilogue: add bias, route into Qs[H][S][64] (x1/8), Ks[H][S][64], Vt[H][64][S]
#pragma unroll
  for (int j = 0; j < 4; ++j) {
    const int colN = tileN * 64 + j * 16 + l16;   // uniform branch per (wave,j)
    const float bv = bias[colN];
#pragma unroll
    for (int mi = 0; mi < 2; ++mi) {
#pragma unroll
      for (int r = 0; r < 8; ++r) {
        const int row = tileM * 32 + mi * 16 + r + half * 8;
        const float v = acc[mi][j][r] + bv;
        if (colN < D_DIM) {
          const int h = colN / W_HD, w = colN % W_HD;
          Qs[((size_t)h * S_LEN + row) * W_HD + w] = (__bf16)(v * 0.125f);
        } else if (colN < 2 * D_DIM) {
          const int nn = colN - D_DIM;
          const int h = nn / W_HD, w = nn % W_HD;
          Ks[((size_t)h * S_LEN + row) * W_HD + w] = (__bf16)v;
        } else {
          const int nn = colN - 2 * D_DIM;
          const int h = nn / W_HD, w = nn % W_HD;
          Vt[((size_t)h * W_HD + w) * S_LEN + row] = (__bf16)v;
        }
      }
    }
  }
}

// ---------- flash attention ----------
// One wave: 16 query rows of one head, 64-key chunks.
__global__ void __launch_bounds__(128)
attn_kernel(const __bf16* __restrict__ Qs, const __bf16* __restrict__ Ks,
            const __bf16* __restrict__ Vt, const float* __restrict__ maskadd,
            float* __restrict__ out)
{
  __shared__ __align__(16) __bf16 plds[4][16][72];   // 16x64 P tile per wave, padded

  const int lane  = threadIdx.x & 31;
  const int wid   = threadIdx.x >> 5;
  const int gw    = blockIdx.x * 4 + wid;
  const int head  = gw >> 8;          // 256 q-tiles per head
  const int qtile = gw & 255;
  const int half  = lane >> 4;
  const int l16   = lane & 15;
  const int q0    = qtile * 16;
  const int aoff  = half * 8;
  const int boff  = half * 8;

  const __bf16* Qh = Qs + (size_t)head * S_LEN * W_HD;
  const __bf16* Kh = Ks + (size_t)head * S_LEN * W_HD;
  const __bf16* Vh = Vt + (size_t)head * W_HD * S_LEN;

  // Q fragments (A layout, K=0..31 and K=32..63), loaded once.
  const __bf16* qrow = Qh + (size_t)(q0 + l16) * W_HD;
  const v16bf aq0 = load_frag16(qrow + aoff,      qrow + 16 + aoff);
  const v16bf aq1 = load_frag16(qrow + 32 + aoff, qrow + 48 + aoff);

  float mrow[8], lrow[8];
  v8f o[4] = {{}, {}, {}, {}};
#pragma unroll
  for (int r = 0; r < 8; ++r) { mrow[r] = -1e30f; lrow[r] = 0.0f; }

  for (int t = 0; t < S_LEN; t += 64) {
    if (t + 64 < S_LEN) {
      __builtin_prefetch(Kh + (size_t)(t + 64 + l16) * W_HD, 0, 0);
      __builtin_prefetch(Kh + (size_t)(t + 96 + l16) * W_HD, 0, 0);
      __builtin_prefetch(Vh + (size_t)l16 * S_LEN + t + 64, 0, 0);
    }

    // ---- scores: four 16x16 tiles (keys t+0..63), K-dim 64 ----
    v8f s[4] = {{}, {}, {}, {}};
#pragma unroll
    for (int kt = 0; kt < 4; ++kt) {
      const __bf16* krow = Kh + (size_t)(t + kt * 16 + l16) * W_HD;
      v16bf bk = load_frag16(krow + boff, krow + boff + 16);
      s[kt] = WMMA_BF16(aq0, bk, s[kt]);
      bk = load_frag16(krow + 32 + boff, krow + 32 + boff + 16);
      s[kt] = WMMA_BF16(aq1, bk, s[kt]);
    }

    const float ma0 = maskadd[t + l16];
    const float ma1 = maskadd[t + 16 + l16];
    const float ma2 = maskadd[t + 32 + l16];
    const float ma3 = maskadd[t + 48 + l16];

    // ---- online softmax over these 64 keys ----
#pragma unroll
    for (int r = 0; r < 8; ++r) {
      const float a0 = s[0][r] + ma0;
      const float a1 = s[1][r] + ma1;
      const float a2 = s[2][r] + ma2;
      const float a3 = s[3][r] + ma3;
      const float tm = rmax16(fmaxf(fmaxf(a0, a1), fmaxf(a2, a3)));
      const float mn = fmaxf(mrow[r], tm);
      const float es = __expf(mrow[r] - mn);
      const float p0 = __expf(a0 - mn);
      const float p1 = __expf(a1 - mn);
      const float p2 = __expf(a2 - mn);
      const float p3 = __expf(a3 - mn);
      mrow[r] = mn;
      lrow[r] = lrow[r] * es + rsum16(p0 + p1 + p2 + p3);
      o[0][r] *= es; o[1][r] *= es; o[2][r] *= es; o[3][r] *= es;
      // stash P (16x64) in LDS, C-layout -> row-major
      const int pr = r + half * 8;
      plds[wid][pr][l16]      = (__bf16)p0;
      plds[wid][pr][16 + l16] = (__bf16)p1;
      plds[wid][pr][32 + l16] = (__bf16)p2;
      plds[wid][pr][48 + l16] = (__bf16)p3;
    }
    asm volatile("s_wait_dscnt 0" ::: "memory");

    // reload P as two A fragments (16x32 each)
    const __bf16* prow = &plds[wid][l16][0];
    const v16bf ap0 = load_frag16(prow + aoff,      prow + 16 + aoff);
    const v16bf ap1 = load_frag16(prow + 32 + aoff, prow + 48 + aoff);

    // ---- O += P @ V : 8 WMMAs, V transposed so B-frags are contiguous ----
#pragma unroll
    for (int wc = 0; wc < 4; ++wc) {
      const __bf16* vrow = Vh + (size_t)(wc * 16 + l16) * S_LEN + t;
      v16bf bv = load_frag16(vrow + boff, vrow + boff + 16);
      o[wc] = WMMA_BF16(ap0, bv, o[wc]);
      bv = load_frag16(vrow + 32 + boff, vrow + 32 + boff + 16);
      o[wc] = WMMA_BF16(ap1, bv, o[wc]);
    }
  }

  // ---- finalize: divide by row sums, write fp32 output (B,S,D) ----
#pragma unroll
  for (int wc = 0; wc < 4; ++wc) {
#pragma unroll
    for (int r = 0; r < 8; ++r) {
      const int row = q0 + r + half * 8;
      out[(size_t)row * D_DIM + head * W_HD + wc * 16 + l16] = o[wc][r] / lrow[r];
    }
  }
}

// ---------- launch ----------

extern "C" void kernel_launch(void* const* d_in, const int* in_sizes, int n_in,
                              void* d_out, int out_size, void* d_ws, size_t ws_size,
                              hipStream_t stream) {
  (void)in_sizes; (void)n_in; (void)out_size; (void)ws_size;

  const float* x    = (const float*)d_in[0];   // (1,4096,768)
  const float* mask = (const float*)d_in[1];   // (1,4096)
  const float* W    = (const float*)d_in[2];   // (768,2304)
  const float* bias = (const float*)d_in[3];   // (2304,)
  float* out        = (float*)d_out;           // (1,4096,768)

  char* ws = (char*)d_ws;
  size_t off = 0;
  auto alloc = [&](size_t bytes) -> void* {
    void* p = ws + off;
    off = (off + bytes + 255) & ~(size_t)255;
    return p;
  };
  __bf16* xb = (__bf16*)alloc((size_t)S_LEN * D_DIM * 2);
  __bf16* wt = (__bf16*)alloc((size_t)D_DIM * QKV3 * 2);
  __bf16* Qs = (__bf16*)alloc((size_t)H_NUM * S_LEN * W_HD * 2);
  __bf16* Ks = (__bf16*)alloc((size_t)H_NUM * S_LEN * W_HD * 2);
  __bf16* Vt = (__bf16*)alloc((size_t)H_NUM * S_LEN * W_HD * 2);
  float*  ma = (float*)alloc((size_t)S_LEN * 4);

  conv_x_kernel<<<(S_LEN * D_DIM + 255) / 256, 256, 0, stream>>>(x, xb, S_LEN * D_DIM);
  conv_w_kernel<<<(D_DIM * QKV3 + 255) / 256, 256, 0, stream>>>(W, wt);
  conv_mask_kernel<<<(S_LEN + 255) / 256, 256, 0, stream>>>(mask, ma);

  // (4096/32) x (2304/64) = 128 x 36 = 4608 tiles, 8 waves per 256-thread block
  qkv_gemm_kernel<<<4608 / 8, 256, 0, stream>>>(xb, wt, bias, Qs, Ks, Vt);

  // 12 heads * 256 q-tiles = 3072 waves, 4 waves per block
  attn_kernel<<<3072 / 4, 128, 0, stream>>>(Qs, Ks, Vt, ma, out);
}
